// GPModule_26474178413382
// MI455X (gfx1250) — matrix operand order repaired and checked by reading, not compiled, not run on hardware
//
#include <hip/hip_runtime.h>
#include <hip/hip_bf16.h>
#include <math.h>

typedef __attribute__((ext_vector_type(16))) __bf16 v16bf;
typedef __attribute__((ext_vector_type(8)))  __bf16 v8bf;
typedef __attribute__((ext_vector_type(8)))  float  v8f;

#define NNODE 512
#define BATCH 32
#define RESC  32
#define SKIPC 64
#define ENDC  128
#define OUTD  12
#define ALPHA_ 0.05f
#define EPS_   1e-5f

// CDNA5 async global->LDS copy. The toolchain's builtin takes
// (v4i addrspace(1)*, v4i addrspace(3)*, imm offset, imm cpol) — signature
// revealed by the round-2 diagnostic. Guarded so absence cannot break the build.
#if defined(__has_builtin)
#  if __has_builtin(__builtin_amdgcn_global_load_async_to_lds_b128)
#    define USE_ASYNC_LDS 1
#  endif
#endif
#ifndef USE_ASYNC_LDS
#  define USE_ASYNC_LDS 0
#endif

#if USE_ASYNC_LDS
typedef int v4i_ __attribute__((vector_size(16)));
typedef __attribute__((address_space(1))) v4i_* gptr128_t;
typedef __attribute__((address_space(3))) v4i_* lptr128_t;
__device__ __forceinline__ void async_copy_b128(const void* gsrc, void* ldst) {
    __builtin_amdgcn_global_load_async_to_lds_b128(
        (gptr128_t)gsrc, (lptr128_t)ldst, 0, 0);
}
__device__ __forceinline__ void wait_async0() {
#if __has_builtin(__builtin_amdgcn_s_wait_asynccnt)
    __builtin_amdgcn_s_wait_asynccnt(0);
#else
    asm volatile("s_wait_asynccnt 0" ::: "memory");
#endif
}
#endif

// ---------------------------------------------------------------------------
// A' = (1-alpha) * rownorm(A + I)   and   A'' = (1-alpha) * rownorm(A^T + I)
// stored as bf16 512x512 (WMMA A-operand). One block per row v.
// ---------------------------------------------------------------------------
__global__ __launch_bounds__(256) void k_prep_A(
    const float* __restrict__ A, __bf16* __restrict__ An, __bf16* __restrict__ AnT)
{
    __shared__ float red[256], redT[256];
    const int v = blockIdx.x, tid = threadIdx.x;
    float sr = 0.f, sc = 0.f;
    for (int w = tid; w < NNODE; w += 256) {
        sr += A[v * NNODE + w];
        sc += A[w * NNODE + v];
    }
    red[tid] = sr; redT[tid] = sc;
    __syncthreads();
    for (int s = 128; s > 0; s >>= 1) {
        if (tid < s) { red[tid] += red[tid + s]; redT[tid] += redT[tid + s]; }
        __syncthreads();
    }
    const float invR = (1.0f - ALPHA_) / (red[0] + 1.0f);
    const float invC = (1.0f - ALPHA_) / (redT[0] + 1.0f);
    for (int w = tid; w < NNODE; w += 256) {
        float d = (w == v) ? 1.f : 0.f;
        An [v * NNODE + w] = (__bf16)((A[v * NNODE + w] + d) * invR);
        AnT[v * NNODE + w] = (__bf16)((A[w * NNODE + v] + d) * invC);
    }
}

// ---------------------------------------------------------------------------
// start conv: 1x1, 2 -> 32 channels.  One thread per (n,v,t), T=64.
// ---------------------------------------------------------------------------
__global__ __launch_bounds__(256) void k_start(
    const float* __restrict__ x, const float* __restrict__ w,
    const float* __restrict__ b, float* __restrict__ X0)
{
    const int p = blockIdx.x * 256 + threadIdx.x;
    const int n = p >> 15;
    const int rem = p & 32767;
    const int v = rem >> 6, t = rem & 63;
    const float i0 = x[((n * 2 + 0) * NNODE + v) * 64 + t];
    const float i1 = x[((n * 2 + 1) * NNODE + v) * 64 + t];
#pragma unroll
    for (int c = 0; c < RESC; ++c)
        X0[(((size_t)n * RESC + c) * NNODE + v) * 64 + t] =
            b[c] + w[c * 2] * i0 + w[c * 2 + 1] * i1;
}

// ---------------------------------------------------------------------------
// generic time-collapsing skip conv: out[n][s][v] (+)= b[s] + sum_{c,t} w*X
// ---------------------------------------------------------------------------
__global__ __launch_bounds__(64) void k_skip(
    float* __restrict__ skip, const float* __restrict__ X,
    const float* __restrict__ w, const float* __restrict__ b,
    int Cin, int L, int accum)
{
    __shared__ float xs[2048];
    const int nv = blockIdx.x;
    const int n = nv >> 9, v = nv & 511;
    const int tot = Cin * L;
    for (int i = threadIdx.x; i < tot; i += 64) {
        int c = i / L, t = i - c * L;
        xs[i] = X[(((size_t)n * Cin + c) * NNODE + v) * L + t];
    }
    __syncthreads();
    const int s = threadIdx.x;
    float acc = b[s];
    const float* wr = w + (size_t)s * tot;
    for (int i = 0; i < tot; ++i) acc += wr[i] * xs[i];
    const size_t o = ((size_t)n * SKIPC + s) * NNODE + v;
    skip[o] = accum ? (skip[o] + acc) : acc;
}

// ---------------------------------------------------------------------------
// fused dilated inception (kernels 2,3,6,7) + tanh/sigmoid gate.
// writes f32 (for mixconv/epilogues) AND bf16 shadow (WMMA B-operand).
// ---------------------------------------------------------------------------
struct IncW {
    const float* fw[4]; const float* fb[4];
    const float* gw[4]; const float* gb[4];
};

__global__ __launch_bounds__(256) void k_inception(
    const float* __restrict__ X, float* __restrict__ xg, __bf16* __restrict__ xgbf,
    IncW P, int Lin, int L, int dil)
{
    __shared__ float Wf[4608], Wg[4608], bF[32], bG[32];
    const int K[4]   = {2, 3, 6, 7};
    const int off[4] = {0, 512, 1280, 2816};
#pragma unroll
    for (int kb = 0; kb < 4; ++kb) {
        int sz = 256 * K[kb];
        for (int i = threadIdx.x; i < sz; i += 256) {
            Wf[off[kb] + i] = P.fw[kb][i];
            Wg[off[kb] + i] = P.gw[kb][i];
        }
        if (threadIdx.x < 8) {
            bF[kb * 8 + threadIdx.x] = P.fb[kb][threadIdx.x];
            bG[kb * 8 + threadIdx.x] = P.gb[kb][threadIdx.x];
        }
    }
    __syncthreads();

    const int p = blockIdx.x * 256 + threadIdx.x;
    const int n = p / (NNODE * L);
    const int rem = p - n * NNODE * L;
    const int v = rem / L, t = rem - (rem / L) * L;

    float accF[32], accG[32];
#pragma unroll
    for (int c = 0; c < 32; ++c) { accF[c] = bF[c]; accG[c] = bG[c]; }

    const float* xb = X + (((size_t)n * RESC) * NNODE + v) * Lin + t;
    const size_t cs = (size_t)NNODE * Lin;

    for (int j = 0; j < 7; ++j) {
        float ch[32];
#pragma unroll
        for (int c = 0; c < 32; ++c) ch[c] = xb[c * cs + dil * j];
#pragma unroll
        for (int kb = 0; kb < 4; ++kb) {
            const int k = K[kb];
            const int i = j - (7 - k);
            if (i < 0) continue;
#pragma unroll
            for (int o = 0; o < 8; ++o) {
                const float* wf = Wf + off[kb] + o * 32 * k + i;
                const float* wg = Wg + off[kb] + o * 32 * k + i;
                float af = 0.f, ag = 0.f;
#pragma unroll
                for (int c = 0; c < 32; ++c) {
                    af += wf[c * k] * ch[c];
                    ag += wg[c * k] * ch[c];
                }
                accF[kb * 8 + o] += af;
                accG[kb * 8 + o] += ag;
            }
        }
    }
    const size_t obase = (((size_t)n * RESC) * NNODE + v) * L + t;
    const size_t cso = (size_t)NNODE * L;
#pragma unroll
    for (int c = 0; c < 32; ++c) {
        float f = tanhf(accF[c]);
        float g = 1.f / (1.f + __expf(-accG[c]));
        float r = f * g;
        xg[obase + c * cso] = r;
        xgbf[obase + c * cso] = (__bf16)r;
    }
}

// ---------------------------------------------------------------------------
// WMMA graph-propagation GEMM:
//   Hout[v,m] = alpha*Xin[v,m] + sum_w A'[v,w]*HinBf[w,m]     (bf16 x bf16 -> f32)
// Block: 512 threads = 16 waves (4x4), block tile 256(V) x 128(M),
// wave tile 64x32 = 4x2 fragments of 16x16, K-step 32.
// Double-buffered LDS, one barrier per K-step; A-tile via async global->LDS.
// ---------------------------------------------------------------------------
#define GSTR 40
__global__ __launch_bounds__(512) void k_gemm_prop(
    float* __restrict__ Hout, __bf16* __restrict__ HoutBf,
    const float* __restrict__ Xin, const __bf16* __restrict__ HinBf,
    const __bf16* __restrict__ Abf, int L, int writeBf)
{
    __shared__ __align__(16) __bf16 As[2][256 * GSTR];
    __shared__ __align__(16) __bf16 Bs[2][128 * GSTR];

    const int tid  = threadIdx.x;
    const int lane = tid & 31;
    const int wave = tid >> 5;
    const int wv = wave >> 2;          // 0..3  (V dir)
    const int wm = wave & 3;           // 0..3  (M dir)
    const int v0 = blockIdx.x * 256;
    const int m0 = blockIdx.y * 128;

    v8f acc[4][2];
#pragma unroll
    for (int a = 0; a < 4; ++a)
#pragma unroll
        for (int b = 0; b < 2; ++b)
#pragma unroll
            for (int e = 0; e < 8; ++e) acc[a][b][e] = 0.f;

    // B addressing: 4 packed-pair elements per thread (bf16 pairs along w)
    const unsigned short* Bu = (const unsigned short*)HinBf;
    size_t bbase[4]; int bml[4], bw2[4];
#pragma unroll
    for (int i = 0; i < 4; ++i) {
        int flat = tid + (i << 9);              // 0..2047
        bml[i] = flat & 127;
        bw2[i] = (flat >> 7) << 1;              // 0,2,..,30
        int m = m0 + bml[i];
        int q = m / L;
        int r = m - q * L;
        bbase[i] = ((size_t)q * NNODE) * (size_t)L + r;
    }
    // A addressing: 2 chunks of 8 bf16 per thread
    int arr[2], akc[2];
#pragma unroll
    for (int i = 0; i < 2; ++i) {
        int flat = tid + (i << 9);              // 0..1023
        arr[i] = flat >> 2;                     // 0..255
        akc[i] = (flat & 3) << 3;               // 0,8,16,24
    }

    // per-lane fragment addressing (ISA 7.12.2 layouts)
    const int arow = lane & 15;
    const int akb  = (lane >> 4) * 8;
    const int bcol = lane & 15;
    const int bkb  = (lane >> 4) * 8;

    // ---- prologue: stage tile 0 ----
#if USE_ASYNC_LDS
#pragma unroll
    for (int i = 0; i < 2; ++i)
        async_copy_b128(Abf + (((size_t)(v0 + arr[i])) << 9) + akc[i],
                        &As[0][arr[i] * GSTR + akc[i]]);
#else
#pragma unroll
    for (int i = 0; i < 2; ++i)
        *(v8bf*)(&As[0][arr[i] * GSTR + akc[i]]) =
            *(const v8bf*)(Abf + (((size_t)(v0 + arr[i])) << 9) + akc[i]);
#endif
#pragma unroll
    for (int i = 0; i < 4; ++i) {
        unsigned lo = Bu[bbase[i] + (size_t)(bw2[i] + 0) * L];
        unsigned hi = Bu[bbase[i] + (size_t)(bw2[i] + 1) * L];
        *(unsigned*)(&Bs[0][bml[i] * GSTR + bw2[i]]) = lo | (hi << 16);
    }
#if USE_ASYNC_LDS
    wait_async0();
#endif
    __syncthreads();

    // ---- main loop: 16 K-steps, double buffered ----
#pragma unroll 2
    for (int it = 0; it < 16; ++it) {
        const int cur = it & 1, nxt = cur ^ 1;
        const int k0n = (it + 1) << 5;
        const bool have = (it < 15);

#if !USE_ASYNC_LDS
        v8bf astage[2];
#endif
        unsigned bstage[4];
        if (have) {
#if USE_ASYNC_LDS
#pragma unroll
            for (int i = 0; i < 2; ++i)
                async_copy_b128(Abf + (((size_t)(v0 + arr[i])) << 9) + k0n + akc[i],
                                &As[nxt][arr[i] * GSTR + akc[i]]);
#else
#pragma unroll
            for (int i = 0; i < 2; ++i)
                astage[i] = *(const v8bf*)
                    (Abf + (((size_t)(v0 + arr[i])) << 9) + k0n + akc[i]);
#endif
#pragma unroll
            for (int i = 0; i < 4; ++i) {
                unsigned lo = Bu[bbase[i] + (size_t)(k0n + bw2[i] + 0) * L];
                unsigned hi = Bu[bbase[i] + (size_t)(k0n + bw2[i] + 1) * L];
                bstage[i] = lo | (hi << 16);
            }
        }

        // fragments from current buffer
        v16bf afr[4], bfr[2];
#pragma unroll
        for (int fa = 0; fa < 4; ++fa) {
            const __bf16* rp = &As[cur][(wv * 64 + fa * 16 + arow) * GSTR + akb];
            v8bf lo = *(const v8bf*)rp;
            v8bf hi = *(const v8bf*)(rp + 16);
            afr[fa] = __builtin_shufflevector(lo, hi,
                0,1,2,3,4,5,6,7,8,9,10,11,12,13,14,15);
        }
#pragma unroll
        for (int fb = 0; fb < 2; ++fb) {
            const __bf16* rp = &Bs[cur][(wm * 32 + fb * 16 + bcol) * GSTR + bkb];
            v8bf lo = *(const v8bf*)rp;
            v8bf hi = *(const v8bf*)(rp + 16);
            bfr[fb] = __builtin_shufflevector(lo, hi,
                0,1,2,3,4,5,6,7,8,9,10,11,12,13,14,15);
        }
#pragma unroll
        for (int fa = 0; fa < 4; ++fa)
#pragma unroll
            for (int fb = 0; fb < 2; ++fb)
                acc[fa][fb] = __builtin_amdgcn_wmma_f32_16x16x32_bf16(
                    false, afr[fa], false, bfr[fb], (short)0, acc[fa][fb], false, false);

        if (have) {
#if !USE_ASYNC_LDS
#pragma unroll
            for (int i = 0; i < 2; ++i)
                *(v8bf*)(&As[nxt][arr[i] * GSTR + akc[i]]) = astage[i];
#endif
#pragma unroll
            for (int i = 0; i < 4; ++i)
                *(unsigned*)(&Bs[nxt][bml[i] * GSTR + bw2[i]]) = bstage[i];
#if USE_ASYNC_LDS
            wait_async0();
#endif
            __syncthreads();
        }
    }

    // epilogue: Hout = alpha*Xin + acc (+ optional bf16 shadow)
    const int vhi = (lane >> 4) << 3;
#pragma unroll
    for (int fb = 0; fb < 2; ++fb) {
        int m = m0 + wm * 32 + fb * 16 + (lane & 15);
        int q = m / L;
        int r = m - q * L;
        size_t mb = ((size_t)q * NNODE) * (size_t)L + r;
#pragma unroll
        for (int fa = 0; fa < 4; ++fa) {
#pragma unroll
            for (int e = 0; e < 8; ++e) {
                int v = v0 + wv * 64 + fa * 16 + e + vhi;
                size_t ix = mb + (size_t)v * L;
                float val = ALPHA_ * Xin[ix] + acc[fa][fb][e];
                Hout[ix] = val;
                if (writeBf) HoutBf[ix] = (__bf16)val;
            }
        }
    }
}

// ---------------------------------------------------------------------------
// 1x1 conv over concat[x,H1,H2] (96 -> 32) with optional accumulate+residual.
// ---------------------------------------------------------------------------
__global__ __launch_bounds__(256) void k_mixconv(
    float* __restrict__ out, const float* __restrict__ x0,
    const float* __restrict__ h1, const float* __restrict__ h2,
    const float* __restrict__ w, const float* __restrict__ b,
    const float* __restrict__ resid, int L, int Lres, int accum)
{
    __shared__ float Wl[32 * 96], bl[32];
    for (int i = threadIdx.x; i < 3072; i += 256) Wl[i] = w[i];
    if (threadIdx.x < 32) bl[threadIdx.x] = b[threadIdx.x];
    __syncthreads();

    const int p = blockIdx.x * 256 + threadIdx.x;
    const int n = p / (NNODE * L);
    const int rem = p - n * NNODE * L;
    const int v = rem / L, l = rem - (rem / L) * L;
    const size_t base = (((size_t)n * RESC) * NNODE + v) * L + l;
    const size_t cs = (size_t)NNODE * L;

    float acc[32];
#pragma unroll
    for (int o = 0; o < 32; ++o) acc[o] = bl[o];
#pragma unroll 4
    for (int c = 0; c < 32; ++c) {
        float xv = x0[base + c * cs];
#pragma unroll
        for (int o = 0; o < 32; ++o) acc[o] += Wl[o * 96 + c] * xv;
    }
#pragma unroll 4
    for (int c = 0; c < 32; ++c) {
        float xv = h1[base + c * cs];
#pragma unroll
        for (int o = 0; o < 32; ++o) acc[o] += Wl[o * 96 + 32 + c] * xv;
    }
#pragma unroll 4
    for (int c = 0; c < 32; ++c) {
        float xv = h2[base + c * cs];
#pragma unroll
        for (int o = 0; o < 32; ++o) acc[o] += Wl[o * 96 + 64 + c] * xv;
    }
    if (accum) {
        const size_t rb = (((size_t)n * RESC) * NNODE + v) * Lres + (l + Lres - L);
        const size_t rcs = (size_t)NNODE * Lres;
#pragma unroll
        for (int o = 0; o < 32; ++o)
            out[base + o * cs] += acc[o] + resid[rb + o * rcs];
    } else {
#pragma unroll
        for (int o = 0; o < 32; ++o) out[base + o * cs] = acc[o];
    }
}

// ---------------------------------------------------------------------------
// layernorm over (C,N,T) per batch sample (biased variance, eps=1e-5)
// ---------------------------------------------------------------------------
__global__ __launch_bounds__(256) void k_lnred(
    const float* __restrict__ X, float* __restrict__ stats, int L)
{
    __shared__ float s1[256], s2[256];
    const int n = blockIdx.x;
    const size_t tot = (size_t)RESC * NNODE * L;
    const float* xb = X + (size_t)n * tot;
    float a = 0.f, bsum = 0.f;
    for (size_t i = threadIdx.x; i < tot; i += 256) {
        float v = xb[i]; a += v; bsum += v * v;
    }
    s1[threadIdx.x] = a; s2[threadIdx.x] = bsum;
    __syncthreads();
    for (int s = 128; s > 0; s >>= 1) {
        if (threadIdx.x < s) {
            s1[threadIdx.x] += s1[threadIdx.x + s];
            s2[threadIdx.x] += s2[threadIdx.x + s];
        }
        __syncthreads();
    }
    if (threadIdx.x == 0) {
        float inv = 1.0f / (float)tot;
        float mu = s1[0] * inv;
        float var = s2[0] * inv - mu * mu;
        stats[n * 2 + 0] = mu;
        stats[n * 2 + 1] = rsqrtf(var + EPS_);
    }
}

__global__ __launch_bounds__(256) void k_lnapply(
    float* __restrict__ X, const float* __restrict__ stats,
    const float* __restrict__ lnw, const float* __restrict__ lnb,
    const int* __restrict__ idx, int L)
{
    const size_t p = (size_t)blockIdx.x * 256 + threadIdx.x;
    const int per = RESC * NNODE * L;
    const int n = (int)(p / per);
    int rem = (int)(p - (size_t)n * per);
    const int c = rem / (NNODE * L);
    rem -= c * NNODE * L;
    const int v = rem / L, l = rem - (rem / L) * L;
    const float mu = stats[n * 2], rstd = stats[n * 2 + 1];
    const size_t wix = ((size_t)c * NNODE + idx[v]) * L + l;
    X[p] = (X[p] - mu) * rstd * lnw[wix] + lnb[wix];
}

// ---------------------------------------------------------------------------
// final: relu(skip) -> end1 (128x64) + relu -> end2 (12x128) -> out (B,12,N,1)
// ---------------------------------------------------------------------------
__global__ __launch_bounds__(256) void k_final(
    const float* __restrict__ skip,
    const float* __restrict__ w1, const float* __restrict__ b1,
    const float* __restrict__ w2, const float* __restrict__ b2,
    float* __restrict__ out)
{
    __shared__ float W1[ENDC * SKIPC];
    __shared__ float W2[OUTD * ENDC];
    __shared__ float B1[ENDC], B2[OUTD];
    for (int i = threadIdx.x; i < ENDC * SKIPC; i += 256) W1[i] = w1[i];
    for (int i = threadIdx.x; i < OUTD * ENDC; i += 256) W2[i] = w2[i];
    if (threadIdx.x < ENDC) B1[threadIdx.x] = b1[threadIdx.x];
    if (threadIdx.x < OUTD) B2[threadIdx.x] = b2[threadIdx.x];
    __syncthreads();

    const int p = blockIdx.x * 256 + threadIdx.x;
    const int n = p >> 9, v = p & 511;
    float s[SKIPC];
#pragma unroll
    for (int k = 0; k < SKIPC; ++k) {
        float t = skip[((size_t)n * SKIPC + k) * NNODE + v];
        s[k] = t > 0.f ? t : 0.f;
    }
    float o12[OUTD];
#pragma unroll
    for (int o = 0; o < OUTD; ++o) o12[o] = B2[o];
    for (int e = 0; e < ENDC; ++e) {
        float acc = B1[e];
#pragma unroll
        for (int k = 0; k < SKIPC; ++k) acc += W1[e * SKIPC + k] * s[k];
        acc = acc > 0.f ? acc : 0.f;
#pragma unroll
        for (int o = 0; o < OUTD; ++o) o12[o] += W2[o * ENDC + e] * acc;
    }
#pragma unroll
    for (int o = 0; o < OUTD; ++o)
        out[((size_t)n * OUTD + o) * NNODE + v] = o12[o];
}

// ---------------------------------------------------------------------------
// host-side orchestration (jax-pytree leaf order: sorted dict keys)
// ---------------------------------------------------------------------------
extern "C" void kernel_launch(void* const* d_in, const int* in_sizes, int n_in,
                              void* d_out, int out_size, void* d_ws, size_t ws_size,
                              hipStream_t stream)
{
    (void)in_sizes; (void)n_in; (void)out_size; (void)ws_size;
    const float* x       = (const float*)d_in[0];
    const float* Amat    = (const float*)d_in[1];
    const int*   idx     = (const int*)d_in[2];
    const float* end1_b  = (const float*)d_in[3];
    const float* end1_w  = (const float*)d_in[4];
    const float* end2_b  = (const float*)d_in[5];
    const float* end2_w  = (const float*)d_in[6];
    const float* skip0_b = (const float*)d_in[79];
    const float* skip0_w = (const float*)d_in[80];
    const float* skipE_b = (const float*)d_in[81];
    const float* skipE_w = (const float*)d_in[82];
    const float* start_b = (const float*)d_in[83];
    const float* start_w = (const float*)d_in[84];

    char* ws = (char*)d_ws;
    __bf16* An  = (__bf16*)(ws);
    __bf16* AnT = (__bf16*)(ws + (size_t)512 * 512 * 2);
    float*  skip  = (float*)(ws + (size_t)2 * 512 * 512 * 2);
    float*  stats = (float*)(ws + (size_t)2 * 512 * 512 * 2 + (size_t)BATCH * SKIPC * NNODE * 4);
    char*   bigbase = ws + (size_t)2 * 512 * 512 * 2 + (size_t)BATCH * SKIPC * NNODE * 4 + 1024;
    const size_t R = (size_t)BATCH * RESC * NNODE * 64 * sizeof(float);   // 128 MB
    float*  big0 = (float*)(bigbase);
    float*  xg   = (float*)(bigbase + R);
    float*  H1   = (float*)(bigbase + 2 * R);
    float*  H2   = (float*)(bigbase + 3 * R);
    float*  big4 = (float*)(bigbase + 4 * R);
    __bf16* xgbf = (__bf16*)(bigbase + 5 * R);
    __bf16* H1bf = (__bf16*)(bigbase + 5 * R + R / 2);

    k_prep_A<<<512, 256, 0, stream>>>(Amat, An, AnT);
    k_start<<<(BATCH * NNODE * 64) / 256, 256, 0, stream>>>(x, start_w, start_b, big0);
    k_skip<<<BATCH * NNODE, 64, 0, stream>>>(skip, x, skip0_w, skip0_b, 2, 64, 0);

    const int TJ[3]   = {58, 46, 22};
    const int DILS[3] = {1, 2, 4};
    float* Xcur  = big0;
    float* Spare = big4;
    for (int j = 0; j < 3; ++j) {
        const int Lin = (j == 0) ? 64 : TJ[j - 1];
        const int L = TJ[j];
        const int d = DILS[j];
        const int base = 7 + 24 * j;
        IncW P;
        for (int k = 0; k < 4; ++k) {
            P.fb[k] = (const float*)d_in[base + k];
            P.fw[k] = (const float*)d_in[base + 4 + k];
            P.gb[k] = (const float*)d_in[base + 8 + k];
            P.gw[k] = (const float*)d_in[base + 12 + k];
        }
        const float* lnb = (const float*)d_in[base + 16];
        const float* lnw = (const float*)d_in[base + 17];
        const float* m1b = (const float*)d_in[base + 18];
        const float* m1w = (const float*)d_in[base + 19];
        const float* m2b = (const float*)d_in[base + 20];
        const float* m2w = (const float*)d_in[base + 21];
        const float* sb  = (const float*)d_in[base + 22];
        const float* sw  = (const float*)d_in[base + 23];

        const int pts = BATCH * NNODE * L;
        k_inception<<<pts / 256, 256, 0, stream>>>(Xcur, xg, xgbf, P, Lin, L, d);
        k_skip<<<BATCH * NNODE, 64, 0, stream>>>(skip, xg, sw, sb, RESC, L, 1);

        const int M = BATCH * RESC * L;          // multiple of 128
        dim3 gg(NNODE / 256, M / 128);
        k_gemm_prop<<<gg, 512, 0, stream>>>(H1, H1bf, xg, xgbf, An, L, 1);
        k_gemm_prop<<<gg, 512, 0, stream>>>(H2, nullptr, xg, H1bf, An, L, 0);
        k_mixconv<<<pts / 256, 256, 0, stream>>>(Spare, xg, H1, H2, m1w, m1b,
                                                 nullptr, L, Lin, 0);
        k_gemm_prop<<<gg, 512, 0, stream>>>(H1, H1bf, xg, xgbf, AnT, L, 1);
        k_gemm_prop<<<gg, 512, 0, stream>>>(H2, nullptr, xg, H1bf, AnT, L, 0);
        k_mixconv<<<pts / 256, 256, 0, stream>>>(Spare, xg, H1, H2, m2w, m2b,
                                                 Xcur, L, Lin, 1);
        k_lnred<<<BATCH, 256, 0, stream>>>(Spare, stats, L);
        const size_t tot = (size_t)BATCH * RESC * NNODE * L;
        k_lnapply<<<(unsigned)(tot / 256), 256, 0, stream>>>(Spare, stats, lnw, lnb, idx, L);
        float* t = Xcur; Xcur = Spare; Spare = t;
    }
    k_skip<<<BATCH * NNODE, 64, 0, stream>>>(skip, Xcur, skipE_w, skipE_b, RESC, 22, 1);
    k_final<<<(BATCH * NNODE) / 256, 256, 0, stream>>>(skip, end1_w, end1_b,
                                                       end2_w, end2_b, (float*)d_out);
}
